// GATLayer_25262997635750
// MI455X (gfx1250) — matrix operand order repaired and checked by reading, not compiled, not run on hardware
//
#include <hip/hip_runtime.h>
#include <stdint.h>

// ---------------------------------------------------------------------------
// GAT layer for MI455X (gfx1250, wave32, WMMA + TDM)
//   h = x @ W                    -> v_wmma_f32_16x16x4_f32 (f32 precision)
//   f1/f2 = h . a1 / h . a2      -> wave32 shuffle reduction
//   att = softmax(mask(lrelu))   -> fused, adj streamed twice (268 MB floor)
//   out = elu(att @ h)           -> v_wmma_f32_16x16x32_bf16, f32 accum,
//                                   adj tiles staged LDS via TENSOR_LOAD_TO_LDS
// ---------------------------------------------------------------------------

typedef __attribute__((ext_vector_type(2)))  float        v2f;
typedef __attribute__((ext_vector_type(8)))  float        v8f;
typedef __attribute__((ext_vector_type(16))) __bf16       v16bf;
typedef __attribute__((ext_vector_type(4)))  unsigned int v4u;
typedef __attribute__((ext_vector_type(4)))  int          v4i;
typedef __attribute__((ext_vector_type(8)))  int          v8i;

#define B_    8
#define M_    2048
#define FIN_  256
#define FOUT_ 128
#define ALPHA 0.2f
#define NEGINF -9.0e15f

__device__ __forceinline__ unsigned short f32_to_bf16(float f) {
  unsigned u = __float_as_uint(f);
  u += 0x7FFFu + ((u >> 16) & 1u);         // round to nearest even
  return (unsigned short)(u >> 16);
}

// ---------------------------------------------------------------------------
// TDM: load one 16x32 int32 tile (row stride 2048 elems) from global -> LDS.
// D# per CDNA5 ISA ch.8: group0 = {count, lds_addr, global_addr, type=2},
// group1 = {data_size=4B, tensor dims, tile dims, dim0 stride}; 2-D tensor so
// groups 2/3 are zero. X-fastest fill => contiguous [row][col] block in LDS.
// ---------------------------------------------------------------------------
__device__ __forceinline__ void tdm_load_adj_tile(const int* gaddr, unsigned lds_off) {
  const unsigned long long ga = (unsigned long long)(uintptr_t)gaddr;
  v4u g0;
  g0[0] = 1u;                                               // count=1 (valid user D#)
  g0[1] = lds_off;                                          // LDS byte address
  g0[2] = (unsigned)ga;                                     // global_addr[31:0]
  g0[3] = ((unsigned)(ga >> 32) & 0x01FFFFFFu) | (2u << 30);// global_addr[56:32] | type=2
  v8i g1;
  g1[0] = 0x00020000;                                       // data_size=4B, wg_mask=0
  g1[1] = (int)(((unsigned)(M_ & 0xFFFF)) << 16);           // tensor_dim0 = 2048 (lo16)
  g1[2] = (int)(((unsigned)(M_ & 0xFFFF)) << 16);           // tensor_dim1 = 2048 (lo16)
  g1[3] = 32 << 16;                                         // tile_dim0 = 32
  g1[4] = 16;                                               // tile_dim1 = 16, tile_dim2 = 0
  g1[5] = M_;                                               // tensor_dim0_stride = 2048
  g1[6] = 0;
  g1[7] = 0;
  const v4i z4 = {0, 0, 0, 0};
#if __clang_major__ >= 23
  const v8i z8 = {0, 0, 0, 0, 0, 0, 0, 0};
  __builtin_amdgcn_tensor_load_to_lds(g0, g1, z4, z4, z8, 0);
#else
  __builtin_amdgcn_tensor_load_to_lds(g0, g1, z4, z4, 0);
#endif
}

// ---------------------------------------------------------------------------
// Kernel 1: h = x @ W   (f32 WMMA 16x16x4)
// One wave per 16-row x 128-col tile of h. 1024 waves total.
// A layout (16x4 f32): lane -> M = lane&15, VGPR v -> K = v + 2*(lane>=16)
// B layout (4x16 f32): lane -> N = lane&15, VGPR v -> K = v + 2*(lane>=16)
// ---------------------------------------------------------------------------
__global__ __launch_bounds__(32)
void k_gemm_xw(const float* __restrict__ x, const float* __restrict__ W,
               float* __restrict__ h) {
  const int lane = threadIdx.x;
  const int lo = lane & 15, hi = lane >> 4;
  const int row0 = blockIdx.x * 16;
  const float* xr = x + (size_t)(row0 + lo) * FIN_;

  v8f acc[8];
  #pragma unroll
  for (int f = 0; f < 8; ++f)
    #pragma unroll
    for (int r = 0; r < 8; ++r) acc[f][r] = 0.0f;

  for (int k = 0; k < FIN_; k += 4) {
    const int ka = k + 2 * hi;
    v2f A;
    A.x = xr[ka];
    A.y = xr[ka + 1];
    #pragma unroll
    for (int f = 0; f < 8; ++f) {
      v2f Bv;
      Bv.x = W[(size_t)ka * FOUT_ + f * 16 + lo];
      Bv.y = W[(size_t)(ka + 1) * FOUT_ + f * 16 + lo];
      acc[f] = __builtin_amdgcn_wmma_f32_16x16x4_f32(
          false, A, false, Bv, (short)0, acc[f], false, false);
    }
  }
  // D layout: VGPR r -> M = r + 8*(lane>=16), N = lane&15
  #pragma unroll
  for (int f = 0; f < 8; ++f)
    #pragma unroll
    for (int r = 0; r < 8; ++r)
      h[(size_t)(row0 + r + 8 * hi) * FOUT_ + f * 16 + lo] = acc[f][r];
}

// ---------------------------------------------------------------------------
// Kernel 2: f1[row] = h[row,:].a1, f2[row] = h[row,:].a2  (wave per row)
// ---------------------------------------------------------------------------
__global__ __launch_bounds__(256)
void k_f1f2(const float* __restrict__ h, const float* __restrict__ a,
            float* __restrict__ f1, float* __restrict__ f2) {
  const int lane = threadIdx.x & 31;
  const int row = blockIdx.x * 8 + (threadIdx.x >> 5);
  const float4 hv  = *(const float4*)(h + (size_t)row * FOUT_ + lane * 4);
  const float4 a1v = *(const float4*)(a + lane * 4);
  const float4 a2v = *(const float4*)(a + FOUT_ + lane * 4);
  float s1 = hv.x * a1v.x + hv.y * a1v.y + hv.z * a1v.z + hv.w * a1v.w;
  float s2 = hv.x * a2v.x + hv.y * a2v.y + hv.z * a2v.z + hv.w * a2v.w;
  #pragma unroll
  for (int off = 16; off > 0; off >>= 1) {
    s1 += __shfl_xor(s1, off, 32);
    s2 += __shfl_xor(s2, off, 32);
  }
  if (lane == 0) { f1[row] = s1; f2[row] = s2; }
}

// ---------------------------------------------------------------------------
// Kernel 3: pre-swizzle h (f32 row-major) into bf16 WMMA B-fragments.
// Fragment id = ((b*64 + chunk)*8 + ntile); fragment = 32 lanes x 32 bytes.
// B layout (32x16 bf16): lane -> N = lane&15, elem e -> K = (e&7) + 8*(lane>=16) + 16*(e>>3)
// ---------------------------------------------------------------------------
__global__ __launch_bounds__(256)
void k_swz(const float* __restrict__ h, unsigned* __restrict__ hswz) {
  const int lane = threadIdx.x & 31;
  const int lo = lane & 15, hi = lane >> 4;
  const int frag = blockIdx.x * 8 + (threadIdx.x >> 5);
  const int t = frag & 7;
  const int c = (frag >> 3) & 63;
  const int b = frag >> 9;
  unsigned pk[8];
  #pragma unroll
  for (int i = 0; i < 8; ++i) {
    const int K0 = 2 * (i & 3) + (hi << 3) + ((i >> 2) << 4);
    const float e0 = h[(size_t)(b * M_ + c * 32 + K0) * FOUT_ + t * 16 + lo];
    const float e1 = h[(size_t)(b * M_ + c * 32 + K0 + 1) * FOUT_ + t * 16 + lo];
    pk[i] = (unsigned)f32_to_bf16(e0) | ((unsigned)f32_to_bf16(e1) << 16);
  }
  uint4* dst = (uint4*)(hswz + (size_t)frag * 256 + lane * 8);
  dst[0] = make_uint4(pk[0], pk[1], pk[2], pk[3]);
  dst[1] = make_uint4(pk[4], pk[5], pk[6], pk[7]);
}

// ---------------------------------------------------------------------------
// Kernel 4: fused masked softmax + attention@h + ELU.
// One block (4 waves) per 16-row tile. Pass 1: waves split rows, online
// (max,sum). Pass 2: waves split the j/K dimension; adj tiles are fetched by
// the Tensor Data Mover into double-buffered LDS (s_wait_tensorcnt pipeline),
// probabilities feed bf16 WMMA 16x16x32; K-split partials merged with
// ds_add_f32 LDS atomics.
// A layout (16x32 bf16): lane -> M = lane&15,
//   elem e=2v+h -> K = 2*(v&3) + h + 8*(lane>=16) + 16*(v>>2)
// ---------------------------------------------------------------------------
__global__ __launch_bounds__(128)
void k_attn(const int* __restrict__ adj, const float* __restrict__ f1g,
            const float* __restrict__ f2g, const unsigned* __restrict__ hswz,
            float* __restrict__ out) {
  __shared__ float f2s[M_];
  __shared__ float accs[16 * FOUT_];
  __shared__ float m_s[16], is_s[16];
  __shared__ int adjblk[4][2][16 * 32];    // per-wave double buffer (TDM dest)

  const int tid = threadIdx.x;
  const int lane = tid & 31;
  const int wv = tid >> 5;
  const int lo = lane & 15, hi = lane >> 4;
  const int b = blockIdx.x >> 7;
  const int i0 = (blockIdx.x & 127) * 16;

  for (int j = tid; j < M_; j += 128) f2s[j] = f2g[b * M_ + j];
  for (int j = tid; j < 16 * FOUT_; j += 128) accs[j] = 0.0f;
  __syncthreads();

  const int* adjb = adj + ((size_t)b * M_ + i0) * M_;

  // ---- pass 1: row-wise online max & exp-sum over all j ----
  for (int r = wv * 4; r < wv * 4 + 4; ++r) {
    const float f1i = f1g[b * M_ + i0 + r];
    const int* arow = adjb + (size_t)r * M_;
    __builtin_prefetch(arow + (M_ / 2) + lane, 0, 0);   // global_prefetch_b8
    float m = NEGINF, s = 0.0f;
    for (int j = lane; j < M_; j += 32) {
      float e = f1i + f2s[j];
      e = (e > 0.0f) ? e : ALPHA * e;
      const float sc = (arow[j] > 0) ? e : NEGINF;
      const float mn = fmaxf(m, sc);
      s = s * __expf(m - mn) + __expf(sc - mn);
      m = mn;
    }
    #pragma unroll
    for (int off = 16; off > 0; off >>= 1) {
      const float mo = __shfl_xor(m, off, 32);
      const float so = __shfl_xor(s, off, 32);
      const float mn = fmaxf(m, mo);
      s = s * __expf(m - mn) + so * __expf(mo - mn);
      m = mn;
    }
    if (lane == 0) { m_s[r] = m; is_s[r] = 1.0f / s; }
  }
  __syncthreads();

  const float mi  = m_s[lo];
  const float isi = is_s[lo];
  const float f1i = f1g[b * M_ + i0 + lo];

  // ---- pass 2: P @ H, waves split the 64 K-chunks; TDM double buffer ----
  v8f acc[8];
  #pragma unroll
  for (int t = 0; t < 8; ++t)
    #pragma unroll
    for (int r = 0; r < 8; ++r) acc[t][r] = 0.0f;

  const unsigned lbuf[2] = {
      (unsigned)(uintptr_t)&adjblk[wv][0][0],
      (unsigned)(uintptr_t)&adjblk[wv][1][0] };

  tdm_load_adj_tile(adjb + wv * 32, lbuf[0]);           // prologue fetch

  int ci = 0;
  for (int c = wv; c < 64; c += 4, ci ^= 1) {
    if (c + 4 < 64) {
      tdm_load_adj_tile(adjb + (c + 4) * 32, lbuf[ci ^ 1]);  // prefetch next
      __builtin_amdgcn_s_wait_tensorcnt(1);                  // current tile ready
    } else {
      __builtin_amdgcn_s_wait_tensorcnt(0);
    }
    const volatile int* blk = &adjblk[wv][ci][0];       // TDM-written; keep after wait
    const int j0 = c * 32;

    union { v16bf v; unsigned short u[16]; } A;
    #pragma unroll
    for (int e = 0; e < 16; ++e) {
      const int v = e >> 1, hh = e & 1;
      const int K = ((v & 3) << 1) + hh + (hi << 3) + ((v >> 2) << 4);
      float ee = f1i + f2s[j0 + K];
      ee = (ee > 0.0f) ? ee : ALPHA * ee;
      const float sc = (blk[lo * 32 + K] > 0) ? ee : NEGINF;
      A.u[e] = f32_to_bf16(__expf(sc - mi) * isi);      // masked -> underflows to 0
    }

    const uint4* hb =
        (const uint4*)(hswz + ((size_t)(b * 64 + c) * 8) * 256) + lane * 2;
    #pragma unroll
    for (int t = 0; t < 8; ++t) {
      union { v16bf v; uint4 q[2]; } Bf;
      Bf.q[0] = hb[t * 64 + 0];     // fragment stride = 256 dwords = 64 uint4
      Bf.q[1] = hb[t * 64 + 1];
      acc[t] = __builtin_amdgcn_wmma_f32_16x16x32_bf16(
          false, A.v, false, Bf.v, (short)0, acc[t], false, false);
    }
  }

  // merge K-split partials across waves (ds_add_f32)
  #pragma unroll
  for (int t = 0; t < 8; ++t)
    #pragma unroll
    for (int r = 0; r < 8; ++r)
      atomicAdd(&accs[(r + 8 * hi) * FOUT_ + t * 16 + lo], acc[t][r]);
  __syncthreads();

  // ELU epilogue
  for (int e = tid; e < 16 * FOUT_; e += 128) {
    const int mrow = e >> 7;
    const int ncol = e & 127;
    const float xv = accs[e];
    out[((size_t)(b * M_ + i0 + mrow)) * FOUT_ + ncol] =
        (xv > 0.0f) ? xv : (__expf(xv) - 1.0f);
  }
}

// ---------------------------------------------------------------------------
// Host launcher
// ---------------------------------------------------------------------------
extern "C" void kernel_launch(void* const* d_in, const int* in_sizes, int n_in,
                              void* d_out, int out_size, void* d_ws, size_t ws_size,
                              hipStream_t stream) {
  (void)in_sizes; (void)n_in; (void)out_size; (void)ws_size;
  const float* x  = (const float*)d_in[0];
  const int*  adj = (const int*)d_in[1];
  const float* W  = (const float*)d_in[2];
  const float* a  = (const float*)d_in[3];
  float* out = (float*)d_out;

  char* ws = (char*)d_ws;
  float*    h    = (float*)ws;                          // 8 MB: h f32
  unsigned* hswz = (unsigned*)(ws + (8u << 20));        // 4 MB: h bf16 swizzled
  float*    f1   = (float*)(ws + (12u << 20));          // 64 KB
  float*    f2   = f1 + B_ * M_;                        // 64 KB

  k_gemm_xw<<<1024, 32, 0, stream>>>(x, W, h);
  k_f1f2  <<<2048, 256, 0, stream>>>(h, a, f1, f2);
  k_swz   <<<512, 256, 0, stream>>>(h, hswz);
  k_attn  <<<1024, 128, 0, stream>>>(adj, f1, f2, hswz, out);
}
